// TemporalAttention_81535659147742
// MI455X (gfx1250) — matrix operand order repaired and checked by reading, not compiled, not run on hardware
//
#include <hip/hip_runtime.h>

// ---------------------------------------------------------------------------
// MultiheadAttention forward for MI455X (gfx1250, wave32, WMMA).
// B=8, S=1024, D=1024, H=16, Dh=64.
// Stage 0: fp32 -> f16 conversion of inputs + weights (bandwidth-trivial).
// Stages 1-3: all GEMMs on v_wmma_f32_16x16x32_f16 with pure-f16 operand
//             loads and ping-pong (copy-free) software pipelining.
// ---------------------------------------------------------------------------

typedef __attribute__((ext_vector_type(16))) _Float16 v16h;
typedef __attribute__((ext_vector_type(8)))  _Float16 v8h;
typedef __attribute__((ext_vector_type(8)))  float    v8f;

#define DEV __device__ __forceinline__

static constexpr int Bsz = 8;
static constexpr int S   = 1024;
static constexpr int D   = 1024;
static constexpr int H   = 16;
static constexpr int Dh  = 64;

DEV v8f wmma_f16(v16h a, v16h b, v8f c) {
  // D = A(16x32,f16) * B(32x16,f16) + C(16x16,f32)
  return __builtin_amdgcn_wmma_f32_16x16x32_f16(
      /*neg_a=*/false, a, /*neg_b=*/false, b,
      /*c_mod=*/(short)0, c, /*reuse_a=*/false, /*reuse_b=*/false);
}

// A operand (16x32 f16), row-major source. row m = lane&15.
// comps 0..7  <- K = k0 + 16g + [0..8)   (one b128 load)
// comps 8..15 <- K = k0 + 16g + 8 + [0..8)
DEV v16h load_a_f16(const _Float16* base, int stride, int m, int k0, int g) {
  v8h lo = *(const v8h*)(base + (size_t)m * stride + k0 + 16 * g);
  v8h hi = *(const v8h*)(base + (size_t)m * stride + k0 + 8 + 16 * g);
  v16h a;
#pragma unroll
  for (int i = 0; i < 8; ++i) { a[i] = lo[i]; a[i + 8] = hi[i]; }
  return a;
}

// B operand (32x16 f16): source is "K-contiguous" per output column n.
// comps c <- K = k0 + 16g + c  (one 32-byte load per lane)
DEV v16h load_b_f16(const _Float16* base, int stride, int n, int k0, int g) {
  return *(const v16h*)(base + (size_t)n * stride + k0 + 16 * g);
}

// ---------------------------------------------------------------------------
// Kernel 0: vectorized fp32 -> f16 conversion (8 elements / thread).
// ---------------------------------------------------------------------------
__global__ __launch_bounds__(256) void cvt_f16_kernel(
    const float* __restrict__ src, _Float16* __restrict__ dst, int n8) {
  const int i = blockIdx.x * blockDim.x + threadIdx.x;
  if (i >= n8) return;
  const float4* s = (const float4*)src + 2 * (size_t)i;
  float4 a = s[0], b = s[1];
  v8h o;
  o[0] = (_Float16)a.x; o[1] = (_Float16)a.y;
  o[2] = (_Float16)a.z; o[3] = (_Float16)a.w;
  o[4] = (_Float16)b.x; o[5] = (_Float16)b.y;
  o[6] = (_Float16)b.z; o[7] = (_Float16)b.w;
  *(v8h*)(dst + 8 * (size_t)i) = o;
}

// ---------------------------------------------------------------------------
// Shared inner pipeline: acc[0..3] += X[16 x K] * W[K x 64]^T tile-product.
// Copy-free ping-pong: buffers 0/1 are alternately loaded (k+32 ahead) and
// consumed by WMMA; no register moves between stages.
// Requires kdim % 64 == 0 and kdim >= 128.
// ---------------------------------------------------------------------------
DEV void gemm_16x64_pipelined(const _Float16* __restrict__ Xrow, int ldx,
                              const _Float16* __restrict__ W, int ldw,
                              int n0, int kdim, int l15, int g, v8f acc[4]) {
  v16h a0, a1, b0[4], b1[4];
  a0 = load_a_f16(Xrow, ldx, l15, 0, g);
#pragma unroll
  for (int nt = 0; nt < 4; ++nt)
    b0[nt] = load_b_f16(W, ldw, n0 + nt * 16 + l15, 0, g);

  for (int k0 = 0; k0 < kdim - 64; k0 += 64) {
    a1 = load_a_f16(Xrow, ldx, l15, k0 + 32, g);
#pragma unroll
    for (int nt = 0; nt < 4; ++nt)
      b1[nt] = load_b_f16(W, ldw, n0 + nt * 16 + l15, k0 + 32, g);
#pragma unroll
    for (int nt = 0; nt < 4; ++nt) acc[nt] = wmma_f16(a0, b0[nt], acc[nt]);

    a0 = load_a_f16(Xrow, ldx, l15, k0 + 64, g);
#pragma unroll
    for (int nt = 0; nt < 4; ++nt)
      b0[nt] = load_b_f16(W, ldw, n0 + nt * 16 + l15, k0 + 64, g);
#pragma unroll
    for (int nt = 0; nt < 4; ++nt) acc[nt] = wmma_f16(a1, b1[nt], acc[nt]);
  }
  // Epilogue: buf0 holds kdim-64; load kdim-32 into buf1.
  a1 = load_a_f16(Xrow, ldx, l15, kdim - 32, g);
#pragma unroll
  for (int nt = 0; nt < 4; ++nt)
    b1[nt] = load_b_f16(W, ldw, n0 + nt * 16 + l15, kdim - 32, g);
#pragma unroll
  for (int nt = 0; nt < 4; ++nt) acc[nt] = wmma_f16(a0, b0[nt], acc[nt]);
#pragma unroll
  for (int nt = 0; nt < 4; ++nt) acc[nt] = wmma_f16(a1, b1[nt], acc[nt]);
}

// ---------------------------------------------------------------------------
// Kernel 1: fused Q/K/V projections (pure-f16 operands).
// out[m,n] = sum_k X[m,k] * W[n,k] + bias[n]    (m = b*S+s, n = h*64+d)
// z=0: Q, scaled by 1/sqrt(Dh)=0.125, layout [B,H,S,Dh]
// z=1: K,                              layout [B,H,S,Dh]
// z=2: V,                              layout [B,H,Dh,S]  (pre-transposed)
// grid: (N/64=16, M/64=128, 3), block: 128 (4 waves; wave w -> 16 rows)
// ---------------------------------------------------------------------------
__global__ __launch_bounds__(128) void qkv_proj_kernel(
    const _Float16* __restrict__ Xq, const _Float16* __restrict__ Xk,
    const _Float16* __restrict__ Xv,
    const _Float16* __restrict__ Wqh, const _Float16* __restrict__ Wkh,
    const _Float16* __restrict__ Wvh,
    const float* __restrict__ bq, const float* __restrict__ bk,
    const float* __restrict__ bv,
    _Float16* __restrict__ Qh, _Float16* __restrict__ Kh,
    _Float16* __restrict__ Vh) {
  const int z = blockIdx.z;
  const _Float16* X    = (z == 0) ? Xq  : (z == 1) ? Xk  : Xv;
  const _Float16* W    = (z == 0) ? Wqh : (z == 1) ? Wkh : Wvh;
  const float*    bias = (z == 0) ? bq  : (z == 1) ? bk  : bv;

  const int lane = threadIdx.x & 31;
  const int wave = threadIdx.x >> 5;
  const int g    = lane >> 4;
  const int l15  = lane & 15;
  const int m0   = blockIdx.y * 64 + wave * 16;
  const int n0   = blockIdx.x * 64;

  v8f acc[4];
#pragma unroll
  for (int nt = 0; nt < 4; ++nt) acc[nt] = (v8f)(0.0f);

  gemm_16x64_pipelined(X + (size_t)m0 * D, D, W, D, n0, D, l15, g, acc);

  const float scale = (z == 0) ? 0.125f : 1.0f;
#pragma unroll
  for (int nt = 0; nt < 4; ++nt) {
    const int n  = n0 + nt * 16 + l15;
    const float bv_ = bias[n];
    const int h = n >> 6, d = n & 63;
#pragma unroll
    for (int r = 0; r < 8; ++r) {
      const int m = m0 + r + 8 * g;       // D-matrix row
      const int b = m >> 10, s = m & 1023;
      const _Float16 hv = (_Float16)((acc[nt][r] + bv_) * scale);
      if (z == 2) {
        Vh[((size_t)(b * H + h) * Dh + d) * S + s] = hv;
      } else {
        _Float16* O = (z == 0) ? Qh : Kh;
        O[((size_t)(b * H + h) * S + s) * Dh + d] = hv;
      }
    }
  }
}

// ---------------------------------------------------------------------------
// Kernel 2: scores + softmax + head-averaged weights + context.
// One workgroup per (b, 16-row q-block). 4 waves loop over 16 heads:
//   scores (WMMA) -> LDS fp32 -> softmax (shuffle reductions) ->
//   P (f16, LDS) + weights accumulation (exclusive global region) ->
//   ctx = P @ V (WMMA, V read pre-transposed from global).
// LDS: 64KB scores + 32KB P = 96KB.
// ---------------------------------------------------------------------------
__global__ __launch_bounds__(128) void attn_kernel(
    const _Float16* __restrict__ Qh, const _Float16* __restrict__ Kh,
    const _Float16* __restrict__ Vh, _Float16* __restrict__ Ctxh,
    float* __restrict__ wgt_out) {
  __shared__ float    sScores[16 * S];
  __shared__ _Float16 sP[16 * S];

  const int b  = blockIdx.y;
  const int q0 = blockIdx.x * 16;
  const int lane = threadIdx.x & 31;
  const int wave = threadIdx.x >> 5;
  const int g    = lane >> 4;
  const int l15  = lane & 15;

  for (int h = 0; h < H; ++h) {
    const _Float16* qptr = Qh + ((size_t)(b * H + h) * S + q0) * Dh;
    const _Float16* kptr = Kh + (size_t)(b * H + h) * S * Dh;
    const _Float16* vtp  = Vh + (size_t)(b * H + h) * Dh * S;

    // Q tile (16 x 64) as two A operands, shared by all 4 waves.
    v16h aq0 = load_a_f16(qptr, Dh, l15, 0, g);
    v16h aq1 = load_a_f16(qptr, Dh, l15, 32, g);

    // --- scores: wave w owns key-columns [w*256, w*256+256) ---
    // Ping-pong over column-tile parity (16 tiles of 16 columns).
    {
      const int cbase = wave * 256 + l15;
      v16h p0a = load_b_f16(kptr, Dh, cbase, 0, g);
      v16h p0b = load_b_f16(kptr, Dh, cbase, 32, g);
      v16h p1a, p1b;
      for (int ct = 0; ct < 14; ct += 2) {
        p1a = load_b_f16(kptr, Dh, cbase + (ct + 1) * 16, 0, g);
        p1b = load_b_f16(kptr, Dh, cbase + (ct + 1) * 16, 32, g);
        {
          v8f acc = (v8f)(0.0f);
          acc = wmma_f16(aq0, p0a, acc);
          acc = wmma_f16(aq1, p0b, acc);
          const int c0 = wave * 256 + ct * 16;
#pragma unroll
          for (int r = 0; r < 8; ++r)
            sScores[(size_t)(r + 8 * g) * S + c0 + l15] = acc[r];
        }
        p0a = load_b_f16(kptr, Dh, cbase + (ct + 2) * 16, 0, g);
        p0b = load_b_f16(kptr, Dh, cbase + (ct + 2) * 16, 32, g);
        {
          v8f acc = (v8f)(0.0f);
          acc = wmma_f16(aq0, p1a, acc);
          acc = wmma_f16(aq1, p1b, acc);
          const int c0 = wave * 256 + (ct + 1) * 16;
#pragma unroll
          for (int r = 0; r < 8; ++r)
            sScores[(size_t)(r + 8 * g) * S + c0 + l15] = acc[r];
        }
      }
      // ct = 14 in buf0, load ct = 15 into buf1.
      p1a = load_b_f16(kptr, Dh, cbase + 15 * 16, 0, g);
      p1b = load_b_f16(kptr, Dh, cbase + 15 * 16, 32, g);
      {
        v8f acc = (v8f)(0.0f);
        acc = wmma_f16(aq0, p0a, acc);
        acc = wmma_f16(aq1, p0b, acc);
        const int c0 = wave * 256 + 14 * 16;
#pragma unroll
        for (int r = 0; r < 8; ++r)
          sScores[(size_t)(r + 8 * g) * S + c0 + l15] = acc[r];
      }
      {
        v8f acc = (v8f)(0.0f);
        acc = wmma_f16(aq0, p1a, acc);
        acc = wmma_f16(aq1, p1b, acc);
        const int c0 = wave * 256 + 15 * 16;
#pragma unroll
        for (int r = 0; r < 8; ++r)
          sScores[(size_t)(r + 8 * g) * S + c0 + l15] = acc[r];
      }
    }
    __syncthreads();

    // --- softmax + weights: wave w owns rows [4w, 4w+4) ---
    for (int rr = 0; rr < 4; ++rr) {
      const int row = wave * 4 + rr;
      float* srow = &sScores[(size_t)row * S];
      float mx = -3.4e38f;
      for (int j = lane; j < S; j += 32) mx = fmaxf(mx, srow[j]);
#pragma unroll
      for (int msk = 16; msk >= 1; msk >>= 1)
        mx = fmaxf(mx, __shfl_xor(mx, msk, 32));
      float sum = 0.0f;
      for (int j = lane; j < S; j += 32) {
        const float e = __expf(srow[j] - mx);
        srow[j] = e;
        sum += e;
      }
#pragma unroll
      for (int msk = 16; msk >= 1; msk >>= 1) sum += __shfl_xor(sum, msk, 32);
      const float inv = 1.0f / sum;
      float* wrow = wgt_out + ((size_t)b * S + q0 + row) * S;
      for (int j = lane; j < S; j += 32) {
        const float p = srow[j] * inv;
        sP[(size_t)row * S + j] = (_Float16)p;
        const float contrib = p * (1.0f / (float)H);
        wrow[j] = (h == 0) ? contrib : (wrow[j] + contrib);  // exclusive region
      }
    }
    __syncthreads();

    // --- ctx = P(16x1024) @ V(1024x64): wave w owns d-cols [16w, 16w+16) ---
    // Ping-pong over K-step parity.
    {
      v8f acc = (v8f)(0.0f);
      const int n = wave * 16 + l15;
      v16h a0 = load_a_f16(sP, S, l15, 0, g);       // LDS (ds_load_b128 x2)
      v16h v0 = load_b_f16(vtp, S, n, 0, g);        // global
      v16h a1, v1;
      for (int k0 = 0; k0 < S - 64; k0 += 64) {
        a1 = load_a_f16(sP, S, l15, k0 + 32, g);
        v1 = load_b_f16(vtp, S, n, k0 + 32, g);
        acc = wmma_f16(a0, v0, acc);
        a0 = load_a_f16(sP, S, l15, k0 + 64, g);
        v0 = load_b_f16(vtp, S, n, k0 + 64, g);
        acc = wmma_f16(a1, v1, acc);
      }
      a1 = load_a_f16(sP, S, l15, S - 32, g);
      v1 = load_b_f16(vtp, S, n, S - 32, g);
      acc = wmma_f16(a0, v0, acc);
      acc = wmma_f16(a1, v1, acc);

      _Float16* cb = Ctxh + ((size_t)b * S + q0) * D + h * Dh;
#pragma unroll
      for (int r = 0; r < 8; ++r)
        cb[(size_t)(r + 8 * g) * D + wave * 16 + l15] = (_Float16)acc[r];
    }
    __syncthreads();
  }
}

// ---------------------------------------------------------------------------
// Kernel 3: output projection. out[m,n] = sum_k Ctx[m,k]*Wo[n,k] + bo[n]
// grid: (16, 128), block 128.
// ---------------------------------------------------------------------------
__global__ __launch_bounds__(128) void out_proj_kernel(
    const _Float16* __restrict__ Ctxh, const _Float16* __restrict__ Woh,
    const float* __restrict__ bo, float* __restrict__ out) {
  const int lane = threadIdx.x & 31;
  const int wave = threadIdx.x >> 5;
  const int g    = lane >> 4;
  const int l15  = lane & 15;
  const int m0   = blockIdx.y * 64 + wave * 16;
  const int n0   = blockIdx.x * 64;

  v8f acc[4];
#pragma unroll
  for (int nt = 0; nt < 4; ++nt) acc[nt] = (v8f)(0.0f);

  gemm_16x64_pipelined(Ctxh + (size_t)m0 * D, D, Woh, D, n0, D, l15, g, acc);

#pragma unroll
  for (int nt = 0; nt < 4; ++nt) {
    const int n = n0 + nt * 16 + l15;
    const float bo_ = bo[n];
#pragma unroll
    for (int r = 0; r < 8; ++r) {
      const int m = m0 + r + 8 * g;
      out[(size_t)m * D + n] = acc[nt][r] + bo_;
    }
  }
}

// ---------------------------------------------------------------------------
extern "C" void kernel_launch(void* const* d_in, const int* in_sizes, int n_in,
                              void* d_out, int out_size, void* d_ws,
                              size_t ws_size, hipStream_t stream) {
  const float* query = (const float*)d_in[0];
  const float* key   = (const float*)d_in[1];
  const float* value = (const float*)d_in[2];
  const float* Wq    = (const float*)d_in[3];
  const float* bq    = (const float*)d_in[4];
  const float* Wk    = (const float*)d_in[5];
  const float* bk    = (const float*)d_in[6];
  const float* Wv    = (const float*)d_in[7];
  const float* bv    = (const float*)d_in[8];
  const float* Wo    = (const float*)d_in[9];
  const float* bo    = (const float*)d_in[10];

  float* out      = (float*)d_out;                       // [B,S,D] attn_output
  float* wgt_out  = out + (size_t)Bsz * S * D;           // [B,S,S] attn_weights

  // Workspace layout (f16):
  //   Qh,Kh,Vh,Ctxh   : 8M halves each (16 MB each)
  //   Xq,Xk,Xv        : 8M halves each (f16 inputs)
  //   Wqh,Wkh,Wvh,Woh : 1M halves each (f16 weights)
  // total = 120 MB
  const size_t BIG = (size_t)Bsz * S * D;                // 8M elements
  const size_t WSZ = (size_t)D * D;                      // 1M elements
  _Float16* Qh   = (_Float16*)d_ws;
  _Float16* Kh   = Qh + BIG;
  _Float16* Vh   = Kh + BIG;
  _Float16* Ctxh = Vh + BIG;
  _Float16* Xq   = Ctxh + BIG;
  _Float16* Xk   = Xq + BIG;
  _Float16* Xv   = Xk + BIG;
  _Float16* Wqh  = Xv + BIG;
  _Float16* Wkh  = Wqh + WSZ;
  _Float16* Wvh  = Wkh + WSZ;
  _Float16* Woh  = Wvh + WSZ;

  // Stage 0: conversions.
  {
    const int n8_big = (int)(BIG / 8);                   // 1M vec8 groups
    const int n8_w   = (int)(WSZ / 8);                   // 128K vec8 groups
    dim3 cb(256);
    dim3 gbig((n8_big + 255) / 256), gw((n8_w + 255) / 256);
    cvt_f16_kernel<<<gbig, cb, 0, stream>>>(query, Xq, n8_big);
    cvt_f16_kernel<<<gbig, cb, 0, stream>>>(key,   Xk, n8_big);
    cvt_f16_kernel<<<gbig, cb, 0, stream>>>(value, Xv, n8_big);
    cvt_f16_kernel<<<gw,   cb, 0, stream>>>(Wq, Wqh, n8_w);
    cvt_f16_kernel<<<gw,   cb, 0, stream>>>(Wk, Wkh, n8_w);
    cvt_f16_kernel<<<gw,   cb, 0, stream>>>(Wv, Wvh, n8_w);
    cvt_f16_kernel<<<gw,   cb, 0, stream>>>(Wo, Woh, n8_w);
  }

  dim3 blk(128);

  dim3 g1(D / 64, (Bsz * S) / 64, 3);                    // 16 x 128 x 3
  qkv_proj_kernel<<<g1, blk, 0, stream>>>(Xq, Xk, Xv, Wqh, Wkh, Wvh,
                                          bq, bk, bv, Qh, Kh, Vh);

  dim3 g2(S / 16, Bsz);                                  // 64 x 8
  attn_kernel<<<g2, blk, 0, stream>>>(Qh, Kh, Vh, Ctxh, wgt_out);

  dim3 g3(D / 64, (Bsz * S) / 64);                       // 16 x 128
  out_proj_kernel<<<g3, blk, 0, stream>>>(Ctxh, Woh, bo, out);
}
